// Decoder_53523882442911
// MI455X (gfx1250) — compile-verified, matrix operand order checked by reading, not run on hardware
//
#include <hip/hip_runtime.h>

// ---------------- problem constants ----------------
#define BB   64      // batch
#define TT   200     // timesteps
#define SS   800     // encoder length
#define HH   512     // hidden
#define CC   512     // context
#define EE   256     // embed
#define VV   34      // vocab
#define KIN  1280    // E + C + H  (gates GEMM K)
#define G4   2048    // 4*H        (gates GEMM N)
#define SCALE_F 0.08838834764831845f  // 1/sqrt(128)

// ---------------- WMMA types ----------------
typedef __attribute__((ext_vector_type(16))) __bf16 bf16x16;
typedef __attribute__((ext_vector_type(8)))  __bf16 bf16x8;
typedef __attribute__((ext_vector_type(8)))  float  f32x8;

static __device__ __forceinline__ f32x8 wmma_bf16(bf16x16 a, bf16x16 b, f32x8 c) {
  // D = A(16x32 bf16) x B(32x16 bf16) + C(16x16 f32)
  return __builtin_amdgcn_wmma_f32_16x16x32_bf16(false, a, false, b, (short)0, c,
                                                 false, false);
}

// Fragment loader for 16-bit 16x32 A (row-major [M][K]) and 32x16 B taken from
// row-major [N][K] storage (i.e. B = W^T). Per the CDNA5 ISA layout:
//   lanes 0-15  : row = lane,    K = 0..7  in v[0..7],  K = 16..23 in v[8..15]
//   lanes 16-31 : row = lane-16, K = 8..15 in v[0..7],  K = 24..31 in v[8..15]
static __device__ __forceinline__ bf16x16 load_frag(const __bf16* base, int ld,
                                                    int row0, int k0) {
  int l  = threadIdx.x & 31;
  int r  = l & 15;
  int kh = (l >> 4) << 3;                       // 0 or 8
  const __bf16* p = base + (size_t)(row0 + r) * ld + k0 + kh;
  bf16x8 lo = *(const bf16x8*)p;                // K = kh .. kh+7
  bf16x8 hi = *(const bf16x8*)(p + 16);         // K = kh+16 .. kh+23
  bf16x16 v;
#pragma unroll
  for (int i = 0; i < 8; ++i) { v[i] = lo[i]; v[i + 8] = hi[i]; }
  return v;
}

static __device__ __forceinline__ float sigmoidf_(float x) {
  return 1.0f / (1.0f + __expf(-x));
}

// ---------------- prep kernels (once per call) ----------------

// Pack Wg = [W_ih | W_hh] -> bf16 [2048][1280] row-major; bg = b_ih + b_hh
__global__ void k_pack_wg(const float* __restrict__ W_ih,
                          const float* __restrict__ W_hh,
                          const float* __restrict__ b_ih,
                          const float* __restrict__ b_hh,
                          __bf16* __restrict__ Wg, float* __restrict__ bg) {
  int idx = blockIdx.x * 256 + threadIdx.x;
  if (idx >= G4 * KIN) return;
  int n = idx / KIN, k = idx % KIN;
  float w = (k < (EE + CC)) ? W_ih[(size_t)n * (EE + CC) + k]
                            : W_hh[(size_t)n * HH + (k - (EE + CC))];
  Wg[idx] = (__bf16)w;
  if (k == 0) bg[n] = b_ih[n] + b_hh[n];
}

__global__ void k_conv_w1(const float* __restrict__ W1, __bf16* __restrict__ W1b) {
  int idx = blockIdx.x * 256 + threadIdx.x;
  if (idx < CC * (HH + CC)) W1b[idx] = (__bf16)W1[idx];
}

// xall[t][b][e] = emb[tokens[b][t]][e]  (bf16)
__global__ void k_embed(const int* __restrict__ tokens,
                        const float* __restrict__ emb,
                        __bf16* __restrict__ xall) {
  int idx = blockIdx.x * 256 + threadIdx.x;
  if (idx >= TT * BB * EE) return;
  int e = idx & (EE - 1);
  int b = (idx >> 8) & (BB - 1);
  int t = idx >> 14;
  int tok = tokens[b * TT + t];
  xall[idx] = (__bf16)emb[(size_t)tok * EE + e];
}

// zero h/c; ac0 = [x_0 | 0 | 0]; ac1 = 0   (acat layout: [x | att | h], K=1280)
__global__ void k_init(const __bf16* __restrict__ xall,
                       float* __restrict__ hbuf, float* __restrict__ cbuf,
                       __bf16* __restrict__ ac0, __bf16* __restrict__ ac1) {
  int idx = blockIdx.x * 256 + threadIdx.x;
  if (idx < BB * HH) { hbuf[idx] = 0.f; cbuf[idx] = 0.f; }
  if (idx < BB * KIN) {
    int row = idx / KIN, col = idx % KIN;
    ac0[idx] = (col < EE) ? xall[row * EE + col] : (__bf16)0.f;
    ac1[idx] = (__bf16)0.f;
  }
}

// ---------------- K1: gates GEMM + fused LSTM cell ----------------
// A = acat_src [64][1280] = [x_t | att | h] (bf16); B = Wg^T (row-major Wg).
// Each wave owns the same 16x16 column tile of i/f/g/o, so the LSTM cell is
// applied elementwise on its own four accumulators. Software-pipelined: the
// next k-tile's 5 fragments are in flight while the current 4 WMMAs issue.
__global__ void __launch_bounds__(128)
k_cell(const __bf16* __restrict__ acat_src, const __bf16* __restrict__ Wg,
       const float* __restrict__ bg, __bf16* __restrict__ acat_dst,
       __bf16* __restrict__ a2, float* __restrict__ hbuf,
       float* __restrict__ cbuf) {
  int wid  = blockIdx.x * 4 + (threadIdx.x >> 5);  // 0..127
  int lane = threadIdx.x & 31;
  int mb = wid >> 5;        // 0..3   (16 batch rows each)
  int jt = wid & 31;        // 0..31  (16 columns of each gate)
  int row0 = mb * 16;
  int nrow = jt * 16;

  f32x8 ai = {}, af = {}, ag = {}, ao = {};

  // prologue: fragments for k-tile 0
  bf16x16 A  = load_frag(acat_src, KIN, row0, 0);
  bf16x16 Bi = load_frag(Wg, KIN, 0 * HH + nrow, 0);
  bf16x16 Bf = load_frag(Wg, KIN, 1 * HH + nrow, 0);
  bf16x16 Bg = load_frag(Wg, KIN, 2 * HH + nrow, 0);
  bf16x16 Bo = load_frag(Wg, KIN, 3 * HH + nrow, 0);

  for (int kt = 0; kt < KIN / 32 - 1; ++kt) {
    int k1 = (kt + 1) * 32;
    // prefetch next tile (independent registers -> partial loadcnt waits)
    bf16x16 An  = load_frag(acat_src, KIN, row0, k1);
    bf16x16 Bin = load_frag(Wg, KIN, 0 * HH + nrow, k1);
    bf16x16 Bfn = load_frag(Wg, KIN, 1 * HH + nrow, k1);
    bf16x16 Bgn = load_frag(Wg, KIN, 2 * HH + nrow, k1);
    bf16x16 Bon = load_frag(Wg, KIN, 3 * HH + nrow, k1);
    ai = wmma_bf16(A, Bi, ai);
    af = wmma_bf16(A, Bf, af);
    ag = wmma_bf16(A, Bg, ag);
    ao = wmma_bf16(A, Bo, ao);
    A = An; Bi = Bin; Bf = Bfn; Bg = Bgn; Bo = Bon;
  }
  ai = wmma_bf16(A, Bi, ai);
  af = wmma_bf16(A, Bf, af);
  ag = wmma_bf16(A, Bg, ag);
  ao = wmma_bf16(A, Bo, ao);

  int ncol = nrow + (lane & 15);
  float bi = bg[0 * HH + ncol], bf_ = bg[1 * HH + ncol];
  float bgv = bg[2 * HH + ncol], bo = bg[3 * HH + ncol];
  int hi = lane >> 4;
#pragma unroll
  for (int e = 0; e < 8; ++e) {
    int m = row0 + e + hi * 8;                 // C/D tile: VGPR e -> M=e / e+8
    size_t idx = (size_t)m * HH + ncol;
    float iv = sigmoidf_(ai[e] + bi);
    float fv = sigmoidf_(af[e] + bf_);
    float gv = tanhf(ag[e] + bgv);
    float ov = sigmoidf_(ao[e] + bo);
    float cn = fv * cbuf[idx] + iv * gv;
    float hn = ov * tanhf(cn);
    cbuf[idx] = cn;
    hbuf[idx] = hn;
    __bf16 hb = (__bf16)hn;
    acat_dst[(size_t)m * KIN + (EE + CC) + ncol] = hb;  // h region of next A
    a2[(size_t)m * (HH + CC) + ncol] = hb;              // [h|ctx] for head GEMM
  }
}

// ---------------- K2: fused attention (energy -> softmax -> context) ----------
// One 512-thread workgroup per batch row; batched GEMV, not WMMA-tileable.
// Also stages x_{t+1} into the destination activation buffer's x-region.
__global__ void __launch_bounds__(512)
k_attn(const float* __restrict__ key, const float* __restrict__ val,
       const int* __restrict__ lens, const float* __restrict__ hbuf,
       const __bf16* __restrict__ xall, __bf16* __restrict__ acat_dst,
       __bf16* __restrict__ a2, int t) {
  __shared__ float ener[SS];
  __shared__ float red[512];
  int b = blockIdx.x, tid = threadIdx.x;
  int lane = tid & 31, wv = tid >> 5;
  int len = lens[b];

  // stage x_{t+1} into dst x-region (read by next step's K1)
  if (t + 1 < TT && tid < EE)
    acat_dst[(size_t)b * KIN + tid] = xall[((size_t)(t + 1) * BB + b) * EE + tid];

  // phase 1: energies. lanes split k (coalesced), wave-reduce per s.
  float hreg[16];
  const float* hp = hbuf + (size_t)b * HH + lane * 16;
#pragma unroll
  for (int i = 0; i < 16; ++i) hreg[i] = hp[i];
  for (int s = wv; s < SS; s += 16) {
    const float* kp = key + ((size_t)b * SS + s) * HH + lane * 16;
    float acc = 0.f;
#pragma unroll
    for (int i = 0; i < 16; ++i) acc += hreg[i] * kp[i];
#pragma unroll
    for (int off = 16; off > 0; off >>= 1) acc += __shfl_xor(acc, off, 32);
    if (lane == 0) ener[s] = (s < len) ? acc * SCALE_F : -__builtin_inff();
  }
  __syncthreads();

  // phase 2: masked softmax stats (max, sum) in LDS
  float mx = -__builtin_inff();
  for (int s = tid; s < SS; s += 512) mx = fmaxf(mx, ener[s]);
  red[tid] = mx; __syncthreads();
  for (int st = 256; st > 0; st >>= 1) {
    if (tid < st) red[tid] = fmaxf(red[tid], red[tid + st]);
    __syncthreads();
  }
  mx = red[0]; __syncthreads();
  float sm = 0.f;
  for (int s = tid; s < SS; s += 512) {
    float e = __expf(ener[s] - mx);  // exp(-inf) = 0 handles mask
    ener[s] = e; sm += e;
  }
  red[tid] = sm; __syncthreads();
  for (int st = 256; st > 0; st >>= 1) {
    if (tid < st) red[tid] += red[tid + st];
    __syncthreads();
  }
  float inv = 1.0f / red[0];

  // phase 3: ctx[c] = inv * sum_s e[s] * value[b][s][c]; threads over c.
  int c = tid;  // 512 threads == CC
  float acc = 0.f;
  for (int s = 0; s < SS; ++s)
    acc += ener[s] * val[((size_t)b * SS + s) * CC + c];
  acc *= inv;
  __bf16 cb = (__bf16)acc;
  acat_dst[(size_t)b * KIN + EE + c] = cb;         // att region of next A
  a2[(size_t)b * (HH + CC) + HH + c] = cb;         // ctx region for head GEMM
}

// ---------------- K3: hid = tanh([h|ctx] @ W1^T + b1), WMMA, pipelined -------
__global__ void __launch_bounds__(128)
k_hid(const __bf16* __restrict__ a2, const __bf16* __restrict__ W1b,
      const float* __restrict__ b1, __bf16* __restrict__ hid) {
  int wid  = blockIdx.x * 4 + (threadIdx.x >> 5);  // 0..127
  int lane = threadIdx.x & 31;
  int mb = wid >> 5, nt = wid & 31;
  f32x8 acc = {};
  bf16x16 A  = load_frag(a2,  HH + CC, mb * 16, 0);
  bf16x16 Bv = load_frag(W1b, HH + CC, nt * 16, 0);
  for (int kt = 0; kt < (HH + CC) / 32 - 1; ++kt) {
    int k1 = (kt + 1) * 32;
    bf16x16 An  = load_frag(a2,  HH + CC, mb * 16, k1);
    bf16x16 Bvn = load_frag(W1b, HH + CC, nt * 16, k1);
    acc = wmma_bf16(A, Bv, acc);
    A = An; Bv = Bvn;
  }
  acc = wmma_bf16(A, Bv, acc);
  int ncol = nt * 16 + (lane & 15);
  float bb = b1[ncol];
  int hi = lane >> 4;
#pragma unroll
  for (int e = 0; e < 8; ++e) {
    int m = mb * 16 + e + hi * 8;
    hid[(size_t)m * CC + ncol] = (__bf16)tanhf(acc[e] + bb);
  }
}

// ---------------- K4: logits = hid @ W2^T + b2 (tiny, VALU) ----------------
__global__ void __launch_bounds__(256)
k_logits(const __bf16* __restrict__ hid, const float* __restrict__ W2,
         const float* __restrict__ b2, float* __restrict__ out, int t) {
  int wid  = blockIdx.x * 8 + (threadIdx.x >> 5);
  int lane = threadIdx.x & 31;
  if (wid >= BB * VV) return;
  int b = wid / VV, v = wid % VV;
  const __bf16* hp = hid + (size_t)b * CC + lane * 16;
  const float*  wp = W2 + (size_t)v * CC + lane * 16;
  float acc = 0.f;
#pragma unroll
  for (int i = 0; i < 16; ++i) acc += (float)hp[i] * wp[i];
#pragma unroll
  for (int off = 16; off > 0; off >>= 1) acc += __shfl_xor(acc, off, 32);
  if (lane == 0) out[((size_t)b * TT + t) * VV + v] = acc + b2[v];
}

// ---------------- workspace layout (bytes, 256-aligned) ----------------
static const size_t OFF_XALL = 0;                         // 200*64*256 bf16
static const size_t OFF_WG   = 6553600;                   // 2048*1280 bf16
static const size_t OFF_BG   = 11796480;                  // 2048 f32
static const size_t OFF_W1   = 11804672;                  // 512*1024 bf16
static const size_t OFF_H    = 12853248;                  // 64*512 f32
static const size_t OFF_C    = 12984320;                  // 64*512 f32
static const size_t OFF_AC0  = 13115392;                  // 64*1280 bf16
static const size_t OFF_AC1  = 13279232;                  // 64*1280 bf16
static const size_t OFF_A2   = 13443072;                  // 64*1024 bf16
static const size_t OFF_HID  = 13574144;                  // 64*512 bf16
static const size_t WS_NEED  = 13639680;

extern "C" void kernel_launch(void* const* d_in, const int* in_sizes, int n_in,
                              void* d_out, int out_size, void* d_ws, size_t ws_size,
                              hipStream_t stream) {
  (void)in_sizes; (void)n_in; (void)out_size;
  if (ws_size < WS_NEED) return;

  const int*   tokens = (const int*)d_in[0];
  const float* key    = (const float*)d_in[1];
  const float* val    = (const float*)d_in[2];
  const int*   lens   = (const int*)d_in[3];
  const float* emb    = (const float*)d_in[4];
  const float* W_ih   = (const float*)d_in[5];
  const float* W_hh   = (const float*)d_in[6];
  const float* b_ih   = (const float*)d_in[7];
  const float* b_hh   = (const float*)d_in[8];
  const float* W1     = (const float*)d_in[9];
  const float* b1     = (const float*)d_in[10];
  const float* W2     = (const float*)d_in[11];
  const float* b2     = (const float*)d_in[12];
  float* out = (float*)d_out;

  char* ws = (char*)d_ws;
  __bf16* xall = (__bf16*)(ws + OFF_XALL);
  __bf16* Wg   = (__bf16*)(ws + OFF_WG);
  float*  bg   = (float*)(ws + OFF_BG);
  __bf16* W1b  = (__bf16*)(ws + OFF_W1);
  float*  hbuf = (float*)(ws + OFF_H);
  float*  cbuf = (float*)(ws + OFF_C);
  __bf16* ac0  = (__bf16*)(ws + OFF_AC0);
  __bf16* ac1  = (__bf16*)(ws + OFF_AC1);
  __bf16* a2   = (__bf16*)(ws + OFF_A2);
  __bf16* hid  = (__bf16*)(ws + OFF_HID);

  // prep (per call; deterministic)
  k_pack_wg<<<(G4 * KIN + 255) / 256, 256, 0, stream>>>(W_ih, W_hh, b_ih, b_hh, Wg, bg);
  k_conv_w1<<<(CC * (HH + CC) + 255) / 256, 256, 0, stream>>>(W1, W1b);
  k_embed<<<(TT * BB * EE + 255) / 256, 256, 0, stream>>>(tokens, emb, xall);
  k_init<<<(BB * KIN + 255) / 256, 256, 0, stream>>>(xall, hbuf, cbuf, ac0, ac1);

  // sequential scan: ping-pong [x|att|h] buffer across steps
  for (int t = 0; t < TT; ++t) {
    __bf16* src = (t & 1) ? ac1 : ac0;
    __bf16* dst = (t & 1) ? ac0 : ac1;
    k_cell<<<32, 128, 0, stream>>>(src, Wg, bg, dst, a2, hbuf, cbuf);
    k_attn<<<BB, 512, 0, stream>>>(key, val, lens, hbuf, xall, dst, a2, t);
    k_hid<<<32, 128, 0, stream>>>(a2, W1b, b1, hid);
    k_logits<<<(BB * VV + 7) / 8, 256, 0, stream>>>(hid, W2, b2, out, t);
  }
}
